// Multi_head_attention_44143673868504
// MI455X (gfx1250) — matrix operand order repaired and checked by reading, not compiled, not run on hardware
//
#include <hip/hip_runtime.h>
#include <cstddef>

typedef __attribute__((ext_vector_type(16))) __bf16 v16bf;
typedef __attribute__((ext_vector_type(8)))  float  v8f;

#define DEV static __device__ __forceinline__

DEV unsigned short f2bf(float f) {
  unsigned int u = __builtin_bit_cast(unsigned int, f);
  u += 0x7FFFu + ((u >> 16) & 1u);  // round-to-nearest-even
  return (unsigned short)(u >> 16);
}

DEV unsigned int pk2(float a, float b) {
  return (unsigned int)f2bf(a) | ((unsigned int)f2bf(b) << 16);
}

// ---- WMMA bf16 16x16x32 fragment loaders (CDNA5 ISA 7.12.2) ----
// A (16 M x 32 K), tile stored M-major: lane row = lane&15, kb = (lane>>4)*8;
// two contiguous 16B runs: K kb+[0..7], kb+16+[0..7].
DEV v16bf load_afrag(const unsigned short* p0, int ld) {
  const int lane = (int)(threadIdx.x & 31u);
  const unsigned short* p = p0 + (lane & 15) * ld + ((lane >> 4) << 3);
  union { uint4 u[2]; v16bf v; } t;
  t.u[0] = *(const uint4*)(p);
  t.u[1] = *(const uint4*)(p + 16);
  return t.v;
}

// B (32 K x 16 N), tile stored TRANSPOSED (n x k): lane col = lane&15,
// kb = (lane>>4)*8; two contiguous 16B runs: K kb+[0..7], kb+16+[0..7].
DEV v16bf load_bfragT(const unsigned short* p0, int ld) {
  const int lane = (int)(threadIdx.x & 31u);
  const unsigned short* p = p0 + (lane & 15) * ld + ((lane >> 4) << 3);
  union { uint4 u[2]; v16bf v; } t;
  t.u[0] = *(const uint4*)(p);
  t.u[1] = *(const uint4*)(p + 16);
  return t.v;
}

DEV v8f wmma_bf16(v16bf a, v16bf b, v8f c) {
  return __builtin_amdgcn_wmma_f32_16x16x32_bf16(false, a, false, b, (short)0, c,
                                                 false, false);
}

// ---------------------------------------------------------------------------
__global__ void cvt_bf16_kernel(const float* __restrict__ in,
                                unsigned short* __restrict__ out, int n) {
  int i = (int)(blockIdx.x * 256u + threadIdx.x);
  if (i < n) out[i] = f2bf(in[i]);
}

// Tiled transpose + f32->bf16: in (R x C) f32 row-major per batch -> out (C x R) bf16.
__global__ __launch_bounds__(256) void cvt_transpose_kernel(
    const float* __restrict__ in, unsigned short* __restrict__ out, int R, int C) {
  __shared__ unsigned short t[64 * 65];
  const size_t ib = (size_t)blockIdx.z * R * C;
  const int r0 = blockIdx.y * 64, c0 = blockIdx.x * 64;
  const int tid = (int)threadIdx.x;
  for (int i = tid; i < 64 * 64; i += 256) {
    const int r = i >> 6, c = i & 63;
    t[r * 65 + c] = f2bf(in[ib + (size_t)(r0 + r) * C + (c0 + c)]);
  }
  __syncthreads();
  for (int i = tid; i < 64 * 64; i += 256) {
    const int c = i >> 6, r = i & 63;  // output-coalesced
    out[ib + (size_t)(c0 + c) * R + (r0 + r)] = t[r * 65 + c];
  }
}

// C = W (1024x1024) @ X + bias per batch; X supplied TRANSPOSED (n-major: L x S).
// Double-buffered LDS, register prefetch, one barrier per k-step.
// OUT_F32=false: write C^T (L x S) bf16, packed 16B stores.
// OUT_F32=true : write C   (S x L) fp32 row-major (final output).
template <bool OUT_F32>
__global__ __launch_bounds__(256, 1) void gemm_bias_kernel(
    const unsigned short* __restrict__ W, const unsigned short* __restrict__ XtAll,
    const float* __restrict__ bias, void* __restrict__ OutAll) {
  constexpr int N = 1024, ITERS = N / 32;
  __shared__ __align__(16) unsigned short sA[2][128 * 32];   // (m, k)
  __shared__ __align__(16) unsigned short sBt[2][128 * 32];  // (n, k)

  const size_t bofs = (size_t)blockIdx.z * N * N;
  const unsigned short* Xt = XtAll + bofs;
  const int n0 = blockIdx.x * 128;
  const int m0 = blockIdx.y * 128;
  const int tid = (int)threadIdx.x;
  const int wave = tid >> 5, lane = tid & 31;
  const int wm = wave & 3, wn = wave >> 2;

  // per-thread staging slot: row r, 16-element half c
  const int r = tid >> 1, c = (tid & 1) * 16;
  const unsigned short* srcA = W + (size_t)(m0 + r) * N + c;
  const unsigned short* srcB = Xt + (size_t)(n0 + r) * N + c;

  v8f acc[2][4];
#pragma unroll
  for (int i = 0; i < 2; ++i)
#pragma unroll
    for (int j = 0; j < 4; ++j) acc[i][j] = {};

  {  // preload k-tile 0 into buffer 0
    *(uint4*)&sA[0][r * 32 + c]      = *(const uint4*)(srcA);
    *(uint4*)&sA[0][r * 32 + c + 8]  = *(const uint4*)(srcA + 8);
    *(uint4*)&sBt[0][r * 32 + c]     = *(const uint4*)(srcB);
    *(uint4*)&sBt[0][r * 32 + c + 8] = *(const uint4*)(srcB + 8);
  }
  __syncthreads();

#pragma unroll 1
  for (int it = 0; it < ITERS; ++it) {
    const int cur = it & 1;
    const bool more = (it + 1) < ITERS;
    uint4 ta0, ta1, tb0, tb1;
    if (more) {  // issue next tile's global loads before compute
      const int k = (it + 1) * 32;
      ta0 = *(const uint4*)(srcA + k);
      ta1 = *(const uint4*)(srcA + k + 8);
      tb0 = *(const uint4*)(srcB + k);
      tb1 = *(const uint4*)(srcB + k + 8);
    }
    v16bf af[2], bfv[4];
#pragma unroll
    for (int i = 0; i < 2; ++i)
      af[i] = load_afrag(&sA[cur][(wm * 32 + i * 16) * 32], 32);
#pragma unroll
    for (int j = 0; j < 4; ++j)
      bfv[j] = load_bfragT(&sBt[cur][(wn * 64 + j * 16) * 32], 32);
#pragma unroll
    for (int i = 0; i < 2; ++i)
#pragma unroll
      for (int j = 0; j < 4; ++j) acc[i][j] = wmma_bf16(af[i], bfv[j], acc[i][j]);
    if (more) {
      const int nxt = 1 - cur;
      *(uint4*)&sA[nxt][r * 32 + c]      = ta0;
      *(uint4*)&sA[nxt][r * 32 + c + 8]  = ta1;
      *(uint4*)&sBt[nxt][r * 32 + c]     = tb0;
      *(uint4*)&sBt[nxt][r * 32 + c + 8] = tb1;
      __syncthreads();
    }
  }

  const int hi = (lane >> 4) * 8;  // D: VGPR r -> row r+hi, col lane&15
#pragma unroll
  for (int i = 0; i < 2; ++i)
#pragma unroll
    for (int j = 0; j < 4; ++j) {
      const int col = n0 + wn * 64 + j * 16 + (lane & 15);
      const int row0 = m0 + wm * 32 + i * 16 + hi;
      if (OUT_F32) {
#pragma unroll
        for (int rr = 0; rr < 8; ++rr)
          ((float*)OutAll)[bofs + (size_t)(row0 + rr) * N + col] =
              acc[i][j][rr] + bias[row0 + rr];
      } else {
        const float4 b0 = *(const float4*)&bias[row0];
        const float4 b1 = *(const float4*)&bias[row0 + 4];
        uint4 q;
        q.x = pk2(acc[i][j][0] + b0.x, acc[i][j][1] + b0.y);
        q.y = pk2(acc[i][j][2] + b0.z, acc[i][j][3] + b0.w);
        q.z = pk2(acc[i][j][4] + b1.x, acc[i][j][5] + b1.y);
        q.w = pk2(acc[i][j][6] + b1.z, acc[i][j][7] + b1.w);
        *(uint4*)&((unsigned short*)OutAll)[bofs + (size_t)col * N + row0] = q;
      }
    }
}

// Pass 1: per head, per score-row l: M[l], Z[l]. Inputs K^T,Q^T (B, L, S).
// Per-lane online (max,sum); double-buffered Q chunks; one cross-lane merge at end.
__global__ __launch_bounds__(256, 1) void softmax_stats_kernel(
    const unsigned short* __restrict__ KtAll, const unsigned short* __restrict__ QtAll,
    float* __restrict__ Mst, float* __restrict__ Zst, float invT) {
  constexpr int L = 1024, S = 1024, D = 64;
  __shared__ __align__(16) unsigned short sKt[128 * 64];     // (l, d)
  __shared__ __align__(16) unsigned short sQt[2][128 * 64];  // (m, d) x2

  const int bh = blockIdx.y, b = bh >> 4, h = bh & 15;
  const size_t hb = (size_t)b * L * S + (size_t)h * D;
  const unsigned short* Kt = KtAll + hb;
  const unsigned short* Qt = QtAll + hb;
  const int l0 = blockIdx.x * 128;
  const int tid = (int)threadIdx.x, wave = tid >> 5, lane = tid & 31;

  for (int i = tid; i < 1024; i += 256) {  // stage K^T block (128x64)
    const int l = i >> 3, cc = (i & 7) * 8;
    *(uint4*)&sKt[l * 64 + cc] = *(const uint4*)&Kt[(size_t)(l0 + l) * S + cc];
  }
  for (int i = tid; i < 1024; i += 256) {  // stage Q chunk 0
    const int m = i >> 3, cc = (i & 7) * 8;
    *(uint4*)&sQt[0][m * 64 + cc] = *(const uint4*)&Qt[(size_t)m * S + cc];
  }
  __syncthreads();
  const v16bf a0 = load_afrag(&sKt[(wave * 16) * 64], 64);
  const v16bf a1 = load_afrag(&sKt[(wave * 16) * 64 + 32], 64);

  float rm[8], rz[8];
#pragma unroll
  for (int r = 0; r < 8; ++r) { rm[r] = -3.0e38f; rz[r] = 0.0f; }

#pragma unroll 1
  for (int ch = 0; ch < 8; ++ch) {  // 8 chunks of 128 m-columns
    const int cur = ch & 1;
    const bool more = ch < 7;
    uint4 pf[4];
    if (more) {  // prefetch next Q chunk into registers
      const int mb = (ch + 1) * 128;
#pragma unroll
      for (int p = 0; p < 4; ++p) {
        const int i = tid + p * 256;
        const int m = i >> 3, cc = (i & 7) * 8;
        pf[p] = *(const uint4*)&Qt[(size_t)(mb + m) * S + cc];
      }
    }
#pragma unroll 1
    for (int mt = 0; mt < 8; ++mt) {  // keep ONE score tile live at a time
      v8f s = {};
      s = wmma_bf16(a0, load_bfragT(&sQt[cur][(mt * 16) * 64], 64), s);
      s = wmma_bf16(a1, load_bfragT(&sQt[cur][(mt * 16) * 64 + 32], 64), s);
#pragma unroll
      for (int r = 0; r < 8; ++r) {  // per-lane online softmax, no shuffles
        const float t = s[r] * invT;
        const float nm = fmaxf(rm[r], t);
        rz[r] = rz[r] * __expf(rm[r] - nm) + __expf(t - nm);
        rm[r] = nm;
      }
    }
    if (more) {
      const int nxt = 1 - cur;
#pragma unroll
      for (int p = 0; p < 4; ++p) {
        const int i = tid + p * 256;
        const int m = i >> 3, cc = (i & 7) * 8;
        *(uint4*)&sQt[nxt][m * 64 + cc] = pf[p];
      }
      __syncthreads();
    }
  }
  // merge (m,z) across the 16 lanes of each half (halves hold different rows)
#pragma unroll
  for (int r = 0; r < 8; ++r) {
#pragma unroll
    for (int msk = 1; msk <= 8; msk <<= 1) {
      const float om = __shfl_xor(rm[r], msk, 32);
      const float oz = __shfl_xor(rz[r], msk, 32);
      const float nm = fmaxf(rm[r], om);
      rz[r] = rz[r] * __expf(rm[r] - nm) + oz * __expf(om - nm);
      rm[r] = nm;
    }
  }
  if ((lane & 15) == 0) {
    const int hi = (lane >> 4) * 8;
#pragma unroll
    for (int r = 0; r < 8; ++r) {
      const int l = l0 + wave * 16 + hi + r;
      Mst[(size_t)bh * L + l] = rm[r];
      Zst[(size_t)bh * L + l] = rz[r];
    }
  }
}

// Pass 2: O[d,m] = sum_l X[d,l] * exp(s[l,m]/T - M[l]) / Z[l]; writes O^T (m, d).
__global__ __launch_bounds__(256, 1) void attn_out_kernel(
    const unsigned short* __restrict__ KtAll, const unsigned short* __restrict__ QtAll,
    const unsigned short* __restrict__ Xall, const float* __restrict__ Mst,
    const float* __restrict__ Zst, unsigned short* __restrict__ AtT, float invT) {
  constexpr int L = 1024, S = 1024, D = 64;
  __shared__ __align__(16) unsigned short sQt[128 * 64];  // (m, d)
  __shared__ __align__(16) unsigned short sKt[32 * 64];   // (l, d)
  __shared__ __align__(16) unsigned short sX[64 * 32];    // (d, l)
  __shared__ __align__(16) unsigned short sPt[128 * 32];  // (m, l) transposed P
  __shared__ float sM[32];
  __shared__ float sZ[32];

  const int bh = blockIdx.y, b = bh >> 4, h = bh & 15;
  const size_t hbT = (size_t)b * L * S + (size_t)h * D;
  const unsigned short* Kt = KtAll + hbT;
  const unsigned short* Qt = QtAll + hbT;
  const unsigned short* X = Xall + (size_t)bh * D * L;  // row-major head (d, l)
  unsigned short* Ot = AtT + hbT;                       // (m, d)

  const int m0 = blockIdx.x * 128;
  const int tid = (int)threadIdx.x, wave = tid >> 5, lane = tid & 31;
  const int wd = wave & 3, wn = wave >> 2;
  const int hi = (lane >> 4) * 8;

  for (int i = tid; i < 1024; i += 256) {  // stage Q^T block once
    const int m = i >> 3, cc = (i & 7) * 8;
    *(uint4*)&sQt[m * 64 + cc] = *(const uint4*)&Qt[(size_t)(m0 + m) * S + cc];
  }

  v8f oacc[4];
#pragma unroll
  for (int j = 0; j < 4; ++j) oacc[j] = {};

#pragma unroll 1
  for (int lt = 0; lt < L; lt += 32) {
    __syncthreads();
    {  // sKt: 32x64, one uint4 per thread
      const int l = tid >> 3, cc = (tid & 7) * 8;
      *(uint4*)&sKt[l * 64 + cc] = *(const uint4*)&Kt[(size_t)(lt + l) * S + cc];
    }
    {  // sX: 64x32, one uint4 per thread
      const int dd = tid >> 2, cc = (tid & 3) * 8;
      *(uint4*)&sX[dd * 32 + cc] = *(const uint4*)&X[(size_t)dd * L + lt + cc];
    }
    if (tid < 32) {
      sM[tid] = Mst[(size_t)bh * L + lt + tid];
      sZ[tid] = 1.0f / Zst[(size_t)bh * L + lt + tid];
    }
    __syncthreads();

    // scores (32 l x 128 m): 16 D-tiles, 2 per wave; emit P^T packed
#pragma unroll 1
    for (int tt = 0; tt < 2; ++tt) {
      const int tile = wave * 2 + tt, tl = tile >> 3, tm = tile & 7;
      v8f s = {};
      s = wmma_bf16(load_afrag(&sKt[(tl * 16) * 64], 64),
                    load_bfragT(&sQt[(tm * 16) * 64], 64), s);
      s = wmma_bf16(load_afrag(&sKt[(tl * 16) * 64 + 32], 64),
                    load_bfragT(&sQt[(tm * 16) * 64 + 32], 64), s);
      const int col = tm * 16 + (lane & 15);
      const int lr0 = tl * 16 + hi;
      float p[8];
#pragma unroll
      for (int r = 0; r < 8; ++r)
        p[r] = __expf(s[r] * invT - sM[lr0 + r]) * sZ[lr0 + r];
      uint4 q;
      q.x = pk2(p[0], p[1]);
      q.y = pk2(p[2], p[3]);
      q.z = pk2(p[4], p[5]);
      q.w = pk2(p[6], p[7]);
      *(uint4*)&sPt[col * 32 + lr0] = q;
    }
    __syncthreads();

    const v16bf ax = load_afrag(&sX[(wd * 16) * 32], 32);
#pragma unroll
    for (int j = 0; j < 4; ++j)
      oacc[j] = wmma_bf16(ax, load_bfragT(&sPt[(wn * 64 + j * 16) * 32], 32), oacc[j]);
  }

#pragma unroll
  for (int j = 0; j < 4; ++j) {  // O^T packed 16B stores
    const int col = m0 + wn * 64 + j * 16 + (lane & 15);
    const int d0 = wd * 16 + hi;
    uint4 q;
    q.x = pk2(oacc[j][0], oacc[j][1]);
    q.y = pk2(oacc[j][2], oacc[j][3]);
    q.z = pk2(oacc[j][4], oacc[j][5]);
    q.w = pk2(oacc[j][6], oacc[j][7]);
    *(uint4*)&Ot[(size_t)col * S + d0] = q;
  }
}

// ---------------------------------------------------------------------------
extern "C" void kernel_launch(void* const* d_in, const int* in_sizes, int n_in,
                              void* d_out, int out_size, void* d_ws, size_t ws_size,
                              hipStream_t stream) {
  (void)in_sizes; (void)n_in; (void)out_size; (void)ws_size;
  constexpr int B = 8, S = 1024, L = 1024, H = 16;

  const float* x  = (const float*)d_in[0];
  const float* y  = (const float*)d_in[1];
  const float* Wk = (const float*)d_in[2];
  const float* bk = (const float*)d_in[3];
  const float* Wq = (const float*)d_in[4];
  const float* bq = (const float*)d_in[5];
  const float* Wp = (const float*)d_in[6];
  const float* bp = (const float*)d_in[7];
  float* out = (float*)d_out;

  const size_t nBSL = (size_t)B * S * L;
  const size_t nSS  = (size_t)S * S;

  char* w = (char*)d_ws;
  auto alloc = [&](size_t bytes) {
    char* p = w;
    w += (bytes + 255) & ~(size_t)255;
    return p;
  };
  unsigned short* xb  = (unsigned short*)alloc(nBSL * 2);  // x  bf16 (B,S,L)
  unsigned short* xt  = (unsigned short*)alloc(nBSL * 2);  // x^T bf16 (B,L,S)
  unsigned short* yt  = (unsigned short*)alloc(nBSL * 2);  // y^T bf16 (B,L,S)
  unsigned short* ktb = (unsigned short*)alloc(nBSL * 2);  // keys^T    (B,L,S)
  unsigned short* qtb = (unsigned short*)alloc(nBSL * 2);  // queries^T (B,L,S)
  unsigned short* atb = (unsigned short*)alloc(nBSL * 2);  // att^T     (B,L,S)
  unsigned short* wkb = (unsigned short*)alloc(nSS * 2);
  unsigned short* wqb = (unsigned short*)alloc(nSS * 2);
  unsigned short* wpb = (unsigned short*)alloc(nSS * 2);
  float* Mst = (float*)alloc((size_t)B * H * L * 4);
  float* Zst = (float*)alloc((size_t)B * H * L * 4);

  cvt_bf16_kernel<<<(unsigned)(nBSL / 256), 256, 0, stream>>>(x, xb, (int)nBSL);
  cvt_bf16_kernel<<<(unsigned)(nSS / 256), 256, 0, stream>>>(Wk, wkb, (int)nSS);
  cvt_bf16_kernel<<<(unsigned)(nSS / 256), 256, 0, stream>>>(Wq, wqb, (int)nSS);
  cvt_bf16_kernel<<<(unsigned)(nSS / 256), 256, 0, stream>>>(Wp, wpb, (int)nSS);

  dim3 gt(L / 64, S / 64, B);
  cvt_transpose_kernel<<<gt, 256, 0, stream>>>(x, xt, S, L);
  cvt_transpose_kernel<<<gt, 256, 0, stream>>>(y, yt, S, L);

  dim3 gg(L / 128, S / 128, B);
  gemm_bias_kernel<false><<<gg, 256, 0, stream>>>(wkb, xt, bk, ktb);  // keys^T
  gemm_bias_kernel<false><<<gg, 256, 0, stream>>>(wqb, yt, bq, qtb);  // queries^T

  const float invT = 2.0f / (float)L;  // scores / (L/2)
  dim3 ga(L / 128, B * H);
  softmax_stats_kernel<<<ga, 256, 0, stream>>>(ktb, qtb, Mst, Zst, invT);
  attn_out_kernel<<<ga, 256, 0, stream>>>(ktb, qtb, xb, Mst, Zst, atb, invT);

  gemm_bias_kernel<true><<<gg, 256, 0, stream>>>(wpb, atb, bp, out);  // projection
}
